// GPT_60181081752195
// MI455X (gfx1250) — compile-verified, matrix operand order checked by reading, not run on hardware
//
#include <hip/hip_runtime.h>
#include <hip/hip_bf16.h>

// ---------------------------------------------------------------------------
// GPT forward for MI455X (gfx1250, wave32, WMMA).
// B=2 S=1024 V=32000 L=6 H=8 DH=64 D=768 FF=2048 DQ=512
// d_out = logits [B*S, V] f32  ++  attns [L,B,H,S,S] f32
// ---------------------------------------------------------------------------

#define B_   2
#define S_   1024
#define V_   32000
#define L_   6
#define H_   8
#define DH_  64
#define D_   768
#define FF_  2048
#define DQ_  512

typedef __attribute__((ext_vector_type(16))) __bf16 v16bf;
typedef __attribute__((ext_vector_type(8)))  __bf16 v8bf;
typedef __attribute__((ext_vector_type(8)))  float  v8f;

#define TM 128     // block M tile
#define TK 32      // K step (one WMMA depth)
#define LDPAD 40   // bf16 LDS row stride (padded; keeps 16B alignment, spreads banks)

__device__ __forceinline__ __bf16 tobf(float x) { return (__bf16)x; }

// ---------------------------------------------------------------------------
// Generic batched strided GEMM: C = alpha * A(f32) x B(f32) [+ bias] [ReLU]
//   A: [M,K] row-major, leading dim lda
//   B: BT==false -> [K,N] (ldb between K rows); BT==true -> [N,K] (ldb between N rows)
//   C: [M,N] f32, leading dim ldc
// Batch z: ptr += (z % batchInner)*sXi + (z / batchInner)*sXo
// f32 -> bf16 conversion happens while staging into LDS.
// Double-buffered LDS: global loads for tile k+1 overlap WMMA on tile k;
// one workgroup barrier per K-step.
// Block = 256 threads = 8 waves (4 M-slices x 2 N-slices); each wave computes
// 32 x (TNT/2) via 2 x (TNT/32) v_wmma_f32_16x16x32_bf16 tiles.
// ---------------------------------------------------------------------------
template<bool BT, int TNT>
__global__ __launch_bounds__(256)
void gemm_wmma(const float* __restrict__ Ag, const float* __restrict__ Bg,
               float* __restrict__ Cg,
               int M, int N, int K,
               int lda, int ldb, int ldc,
               int batchInner,
               long long sAi, long long sAo,
               long long sBi, long long sBo,
               long long sCi, long long sCo,
               float alpha, const float* __restrict__ bias, int relu)
{
    constexpr int NJ   = TNT / 32;   // 16-col WMMA tiles per wave (2 or 4)
    constexpr int EPB  = TNT / 8;    // B elements staged per thread (8 or 16)
    constexpr int TPRB = 256 / TNT;  // BT path: threads per B row (EPB*TPRB == TK)

    __shared__ __bf16 As[2][TM * LDPAD];
    __shared__ __bf16 Bs[2][TNT * LDPAD];

    const int tid      = threadIdx.x;
    const int lane     = tid & 31;
    const int wave     = tid >> 5;
    const int wm       = wave & 3;    // 0..3 : 32-row slice of the 128-row tile
    const int wn       = wave >> 2;   // 0..1 : (TNT/2)-col slice
    const int laneHalf = lane >> 4;   // 0/1  : K-phase / M-half per ISA layout
    const int lane16   = lane & 15;

    const int z  = blockIdx.z;
    const int zi = z % batchInner;
    const int zo = z / batchInner;
    const float* A  = Ag + zi * sAi + zo * sAo;
    const float* Bp = Bg + zi * sBi + zo * sBo;
    float*       C  = Cg + zi * sCi + zo * sCo;

    const int m0 = blockIdx.y * TM;
    const int n0 = blockIdx.x * TNT;

    v8f acc[2][NJ] = {};

    // A staging: 128 rows x 32 k = 16 f32 per thread (4x float4)
    const int ar = tid >> 1;              // 0..127
    const int ac = (tid & 1) * 16;        // 0 / 16
    // B staging indices
    const int brT = tid / TPRB;                 // BT: n row   0..TNT-1
    const int bcT = (tid % TPRB) * EPB;         // BT: k chunk
    const int brN = tid >> 3;                   // NN: k row   0..31
    const int bcN = (tid & 7) * EPB;            // NN: n chunk

    float4 aReg[4];
    float4 bReg[EPB / 4];

    auto loadA = [&](int k0) {
        const float4* src = (const float4*)(A + (long long)(m0 + ar) * lda + k0 + ac);
        #pragma unroll
        for (int j = 0; j < 4; ++j) aReg[j] = src[j];
    };
    auto loadB = [&](int k0) {
        if (BT) {
            const float4* src = (const float4*)(Bp + (long long)(n0 + brT) * ldb + k0 + bcT);
            #pragma unroll
            for (int j = 0; j < EPB / 4; ++j) bReg[j] = src[j];
        } else {
            const float4* src = (const float4*)(Bp + (long long)(k0 + brN) * ldb + n0 + bcN);
            #pragma unroll
            for (int j = 0; j < EPB / 4; ++j) bReg[j] = src[j];
        }
    };
    auto storeA = [&](int buf) {
        __bf16 tmp[16];
        #pragma unroll
        for (int j = 0; j < 4; ++j) {
            tmp[4*j+0] = tobf(aReg[j].x); tmp[4*j+1] = tobf(aReg[j].y);
            tmp[4*j+2] = tobf(aReg[j].z); tmp[4*j+3] = tobf(aReg[j].w);
        }
        v8bf* dst = (v8bf*)&As[buf][ar * LDPAD + ac];
        dst[0] = *(v8bf*)&tmp[0];
        dst[1] = *(v8bf*)&tmp[8];
    };
    auto storeB = [&](int buf) {
        if (BT) {
            // straight copy: Bs[n][k], contiguous in K
            __bf16 tmp[EPB];
            #pragma unroll
            for (int j = 0; j < EPB / 4; ++j) {
                tmp[4*j+0] = tobf(bReg[j].x); tmp[4*j+1] = tobf(bReg[j].y);
                tmp[4*j+2] = tobf(bReg[j].z); tmp[4*j+3] = tobf(bReg[j].w);
            }
            v8bf* dst = (v8bf*)&Bs[buf][brT * LDPAD + bcT];
            #pragma unroll
            for (int j = 0; j < EPB / 8; ++j) dst[j] = *(v8bf*)&tmp[8*j];
        } else {
            // transpose into LDS: read contiguous N, scatter over K
            #pragma unroll
            for (int j = 0; j < EPB / 4; ++j) {
                Bs[buf][(bcN + 4*j + 0) * LDPAD + brN] = tobf(bReg[j].x);
                Bs[buf][(bcN + 4*j + 1) * LDPAD + brN] = tobf(bReg[j].y);
                Bs[buf][(bcN + 4*j + 2) * LDPAD + brN] = tobf(bReg[j].z);
                Bs[buf][(bcN + 4*j + 3) * LDPAD + brN] = tobf(bReg[j].w);
            }
        }
    };
    // Fragments per ISA 16-bit A layout: per lane two 8-element K chunks at
    // k = laneHalf*8 and laneHalf*8 + 16; B staged n-major so it mirrors A.
    auto compute = [&](int buf) {
        v16bf af[2], bfv[NJ];
        #pragma unroll
        for (int i = 0; i < 2; ++i) {
            const __bf16* ra = &As[buf][(wm * 32 + i * 16 + lane16) * LDPAD + laneHalf * 8];
            v8bf lo = *(const v8bf*)(ra);
            v8bf hi = *(const v8bf*)(ra + 16);
            #pragma unroll
            for (int e = 0; e < 8; ++e) { af[i][e] = lo[e]; af[i][8 + e] = hi[e]; }
        }
        #pragma unroll
        for (int j = 0; j < NJ; ++j) {
            const __bf16* rb = &Bs[buf][(wn * (TNT/2) + j * 16 + lane16) * LDPAD + laneHalf * 8];
            v8bf lo = *(const v8bf*)(rb);
            v8bf hi = *(const v8bf*)(rb + 16);
            #pragma unroll
            for (int e = 0; e < 8; ++e) { bfv[j][e] = lo[e]; bfv[j][8 + e] = hi[e]; }
        }
        #pragma unroll
        for (int i = 0; i < 2; ++i)
            #pragma unroll
            for (int j = 0; j < NJ; ++j)
                acc[i][j] = __builtin_amdgcn_wmma_f32_16x16x32_bf16(
                    false, af[i], false, bfv[j], (short)0, acc[i][j], false, false);
    };

    // ---- software pipeline: prologue stage tile 0 --------------------------
    loadA(0); loadB(0);
    storeA(0); storeB(0);
    __syncthreads();

    int cur = 0;
    for (int k0 = TK; k0 < K; k0 += TK) {
        loadA(k0); loadB(k0);     // global loads for tile k0 in flight...
        compute(cur);             // ...while WMMA consumes the current tile
        storeA(cur ^ 1); storeB(cur ^ 1);
        __syncthreads();
        cur ^= 1;
    }
    compute(cur);

    // ---- epilogue: C/D layout -> lane l, vgpr r : m = r + (l/16)*8, n = l%16
    #pragma unroll
    for (int i = 0; i < 2; ++i) {
        #pragma unroll
        for (int j = 0; j < NJ; ++j) {
            const int n  = n0 + wn * (TNT/2) + j * 16 + lane16;
            const float bv = bias ? bias[n] : 0.0f;
            #pragma unroll
            for (int r = 0; r < 8; ++r) {
                const int m = m0 + wm * 32 + i * 16 + laneHalf * 8 + r;
                float vv = acc[i][j][r] * alpha + bv;
                if (relu) vv = fmaxf(vv, 0.0f);
                C[(long long)m * ldc + n] = vv;
            }
        }
    }
}

// ---------------------------------------------------------------------------
// Masked softmax, in place over one attention row of length S.
// grid = (S, B*H). mask: key col > row (causal) OR token[b,col]==0 (pad) -> -1e9
// ---------------------------------------------------------------------------
__global__ __launch_bounds__(256)
void softmax_mask_kernel(float* __restrict__ probs, const int* __restrict__ tokens)
{
    const int row = blockIdx.x;
    const int bh  = blockIdx.y;
    const int b   = bh / H_;
    float* p = probs + (long long)bh * S_ * S_ + (long long)row * S_;
    const int tid = threadIdx.x;
    __shared__ float red[256];

    float vals[4];
    float vmax = -1e30f;
    #pragma unroll
    for (int i = 0; i < 4; ++i) {
        const int c = tid + i * 256;
        float v = p[c];
        const bool masked = (c > row) || (tokens[b * S_ + c] == 0);
        v = masked ? -1e9f : v;
        vals[i] = v;
        vmax = fmaxf(vmax, v);
    }
    red[tid] = vmax; __syncthreads();
    for (int s = 128; s > 0; s >>= 1) {
        if (tid < s) red[tid] = fmaxf(red[tid], red[tid + s]);
        __syncthreads();
    }
    const float m = red[0]; __syncthreads();

    float sum = 0.0f;
    #pragma unroll
    for (int i = 0; i < 4; ++i) { vals[i] = __expf(vals[i] - m); sum += vals[i]; }
    red[tid] = sum; __syncthreads();
    for (int s = 128; s > 0; s >>= 1) {
        if (tid < s) red[tid] += red[tid + s];
        __syncthreads();
    }
    const float inv = 1.0f / red[0];
    #pragma unroll
    for (int i = 0; i < 4; ++i) p[tid + i * 256] = vals[i] * inv;
}

// ---------------------------------------------------------------------------
// h = LayerNorm(h + delta) * g + b   (row length D_=768, 3 elems/thread)
// ---------------------------------------------------------------------------
__global__ __launch_bounds__(256)
void add_ln_kernel(float* __restrict__ h, const float* __restrict__ delta,
                   const float* __restrict__ g, const float* __restrict__ bta)
{
    const int row = blockIdx.x;
    const int tid = threadIdx.x;
    float*       hr = h     + (long long)row * D_;
    const float* dr = delta + (long long)row * D_;
    __shared__ float red[256];

    float x[3];
    float s = 0.0f;
    #pragma unroll
    for (int i = 0; i < 3; ++i) { const int c = tid + i * 256; x[i] = hr[c] + dr[c]; s += x[i]; }
    red[tid] = s; __syncthreads();
    for (int t = 128; t > 0; t >>= 1) { if (tid < t) red[tid] += red[tid + t]; __syncthreads(); }
    const float mean = red[0] * (1.0f / D_); __syncthreads();

    float s2 = 0.0f;
    #pragma unroll
    for (int i = 0; i < 3; ++i) { const float d = x[i] - mean; s2 += d * d; }
    red[tid] = s2; __syncthreads();
    for (int t = 128; t > 0; t >>= 1) { if (tid < t) red[tid] += red[tid + t]; __syncthreads(); }
    const float rs = rsqrtf(red[0] * (1.0f / D_) + 1e-5f);

    #pragma unroll
    for (int i = 0; i < 3; ++i) {
        const int c = tid + i * 256;
        hr[c] = (x[i] - mean) * rs * g[c] + bta[c];
    }
}

// ---------------------------------------------------------------------------
// h[row, :] = word_emb[token[row], :] + pos_emb[row % S, :]
// ---------------------------------------------------------------------------
__global__ __launch_bounds__(256)
void embed_kernel(const int* __restrict__ tokens, const float* __restrict__ we,
                  const float* __restrict__ pe, float* __restrict__ h)
{
    const int row = blockIdx.x;      // b*S + s
    const int s   = row % S_;
    const int tok = tokens[row];
    const int tid = threadIdx.x;
    #pragma unroll
    for (int i = 0; i < 3; ++i) {
        const int c = tid + i * 256;
        h[(long long)row * D_ + c] = we[(long long)tok * D_ + c] + pe[(long long)s * D_ + c];
    }
}

// ---------------------------------------------------------------------------
extern "C" void kernel_launch(void* const* d_in, const int* in_sizes, int n_in,
                              void* d_out, int out_size, void* d_ws, size_t ws_size,
                              hipStream_t stream)
{
    (void)in_sizes; (void)n_in; (void)out_size; (void)ws_size;

    const int*   tokens = (const int*)  d_in[0];
    const float* we     = (const float*)d_in[1];
    const float* pe     = (const float*)d_in[2];
    const float* wq     = (const float*)d_in[3];
    const float* wk     = (const float*)d_in[4];
    const float* wv     = (const float*)d_in[5];
    const float* wo     = (const float*)d_in[6];
    const float* ln1g   = (const float*)d_in[7];
    const float* ln1b   = (const float*)d_in[8];
    const float* w1     = (const float*)d_in[9];
    const float* w2     = (const float*)d_in[10];
    const float* ln2g   = (const float*)d_in[11];
    const float* ln2b   = (const float*)d_in[12];
    const float* pw     = (const float*)d_in[13];
    const float* pb     = (const float*)d_in[14];

    float* logits = (float*)d_out;                               // [2048, 32000]
    float* attns  = logits + (long long)B_ * S_ * V_;            // [L,B,H,S,S]

    // workspace (f32): ~46 MB total
    float* h    = (float*)d_ws;                          // [2048, 768]
    float* q    = h    + (long long)B_ * S_ * D_;        // [2048, 512]
    float* k    = q    + (long long)B_ * S_ * DQ_;       // [2048, 512]
    float* v    = k    + (long long)B_ * S_ * DQ_;       // [2048, 512]
    float* o    = v    + (long long)B_ * S_ * DQ_;       // [2048, 512]
    float* t768 = o    + (long long)B_ * S_ * DQ_;       // [2048, 768]
    float* ff1  = t768 + (long long)B_ * S_ * D_;        // [2048, 2048]

    const int M = B_ * S_;           // 2048
    const dim3 blk(256);
    const long long SS  = (long long)S_ * S_;
    const long long SDQ = (long long)S_ * DQ_;

    // embeddings
    embed_kernel<<<dim3(M), blk, 0, stream>>>(tokens, we, pe, h);

    for (int l = 0; l < L_; ++l) {
        const float* wq_l = wq + (long long)l * D_ * DQ_;
        const float* wk_l = wk + (long long)l * D_ * DQ_;
        const float* wv_l = wv + (long long)l * D_ * DQ_;
        const float* wo_l = wo + (long long)l * DQ_ * D_;
        const float* w1_l = w1 + (long long)l * D_ * FF_;
        const float* w2_l = w2 + (long long)l * FF_ * D_;
        float* attnL = attns + (long long)l * B_ * H_ * S_ * S_;

        // QKV projections: [2048,768] x [768,512] -> [2048,512]
        gemm_wmma<false,128><<<dim3(DQ_/128, M/TM, 1), blk, 0, stream>>>(
            h, wq_l, q, M, DQ_, D_, D_, DQ_, DQ_, 1, 0,0, 0,0, 0,0, 1.0f, nullptr, 0);
        gemm_wmma<false,128><<<dim3(DQ_/128, M/TM, 1), blk, 0, stream>>>(
            h, wk_l, k, M, DQ_, D_, D_, DQ_, DQ_, 1, 0,0, 0,0, 0,0, 1.0f, nullptr, 0);
        gemm_wmma<false,128><<<dim3(DQ_/128, M/TM, 1), blk, 0, stream>>>(
            h, wv_l, v, M, DQ_, D_, D_, DQ_, DQ_, 1, 0,0, 0,0, 0,0, 1.0f, nullptr, 0);

        // scores = (q @ k^T) / sqrt(DH) per (b,head), written directly into attns
        // A = q slice (rows stride 512), B = k slice as [N=S, K=64] (NT)
        gemm_wmma<true,128><<<dim3(S_/128, S_/TM, B_*H_), blk, 0, stream>>>(
            q, k, attnL, S_, S_, DH_, DQ_, DQ_, S_,
            H_, DH_, SDQ, DH_, SDQ, SS, (long long)H_ * SS,
            0.125f, nullptr, 0);

        // masked softmax in place (this IS the attns output)
        softmax_mask_kernel<<<dim3(S_, B_*H_), blk, 0, stream>>>(attnL, tokens);

        // o = probs @ v per (b,head) -> [B,S,H*DH] layout   (N = 64 tile)
        gemm_wmma<false,64><<<dim3(DH_/64, S_/TM, B_*H_), blk, 0, stream>>>(
            attnL, v, o, S_, DH_, S_, S_, DQ_, DQ_,
            H_, SS, (long long)H_ * SS, DH_, SDQ, DH_, SDQ,
            1.0f, nullptr, 0);

        // attention output projection: [2048,512] x [512,768]
        gemm_wmma<false,128><<<dim3(D_/128, M/TM, 1), blk, 0, stream>>>(
            o, wo_l, t768, M, D_, DQ_, DQ_, D_, D_, 1, 0,0, 0,0, 0,0, 1.0f, nullptr, 0);

        // h = LN(h + attn_out)
        add_ln_kernel<<<dim3(M), blk, 0, stream>>>(h, t768, ln1g + l*D_, ln1b + l*D_);

        // FFN: relu(h @ w1) @ w2
        gemm_wmma<false,128><<<dim3(FF_/128, M/TM, 1), blk, 0, stream>>>(
            h, w1_l, ff1, M, FF_, D_, D_, FF_, FF_, 1, 0,0, 0,0, 0,0, 1.0f, nullptr, 1);
        gemm_wmma<false,128><<<dim3(D_/128, M/TM, 1), blk, 0, stream>>>(
            ff1, w2_l, t768, M, D_, FF_, FF_, D_, D_, 1, 0,0, 0,0, 0,0, 1.0f, nullptr, 0);

        // h = LN(h + ffn_out)
        add_ln_kernel<<<dim3(M), blk, 0, stream>>>(h, t768, ln2g + l*D_, ln2b + l*D_);
    }

    // final projection: [2048,768] x [768,32000] + bias -> logits
    gemm_wmma<false,128><<<dim3(V_/128, M/TM, 1), blk, 0, stream>>>(
        h, pw, logits, M, V_, D_, D_, V_, V_, 1, 0,0, 0,0, 0,0, 1.0f, pb, 0);
}
